// Volume_construct_53996328845951
// MI455X (gfx1250) — matrix operand order repaired and checked by reading, not compiled, not run on hardware
//
#include <hip/hip_runtime.h>

// ---------------------------------------------------------------------------
// MI455X (gfx1250) fused cost-volume kernel.
// One workgroup (256 thr = 8 wave32) per (batch, row). All matrix math on
// v_wmma_f32_16x16x32_bf16. Bandwidth-bound: ~341 MB HBM traffic -> ~15us
// floor @ 23.3 TB/s. Round 2: wave-id made SGPR-uniform (no exec-masked
// loops around WMMA), bias hoisted to registers, b128 output writeback.
// ---------------------------------------------------------------------------

typedef __attribute__((ext_vector_type(16))) __bf16 v16bf;
typedef __attribute__((ext_vector_type(8)))  float  v8f;
typedef __attribute__((ext_vector_type(4)))  float  v4f;
typedef __attribute__((ext_vector_type(4)))  unsigned int v4u;
typedef __attribute__((ext_vector_type(8)))  unsigned int v8u;

static_assert(sizeof(v8u) == sizeof(v16bf), "frag size");

__device__ __forceinline__ v16bf frag_cast(v4u a, v4u b) {
    v8u t = __builtin_shufflevector(a, b, 0, 1, 2, 3, 4, 5, 6, 7);
    return __builtin_bit_cast(v16bf, t);
}

__device__ __forceinline__ unsigned short f32_to_bf16(float f) {
    unsigned int u = __float_as_uint(f);
    u += 0x7fffu + ((u >> 16) & 1u);   // round-to-nearest-even
    return (unsigned short)(u >> 16);
}

// Problem dims
#define BB   8
#define CIN  80
#define HH   160
#define WW   320
#define COUT 32
#define DD   48

// LDS strides (bf16 elems unless noted): every fragment chunk is 16B aligned,
// strides spread across the 64 LDS banks.
#define XS   104   // staging [w][c], c padded 80->96 (+pad)
#define CS   40    // feat    [w][c], c = 32 (+pad)
#define VSTR 324   // volume  [i][w] f32 stride (324 % 64 = 4; % 4 == 0 for b128)

// LDS byte offsets (all 16B aligned)
#define OFF_SCALE 0
#define OFF_SHIFT 320
#define OFF_BIAS  640
#define OFF_WB    768                      // [32][XS] bf16 = 6656 B
#define OFF_FL    7424                     // [320][CS] bf16 = 25600 B
#define OFF_FR    33024                    // [320][CS] bf16 = 25600 B
#define OFF_X     58624                    // [320][XS] bf16 = 66560 B (overlaid)
#define SMEM_TOT  125184                   // volume [48][VSTR] f32 = 62208 B fits in X

__global__ void __launch_bounds__(256)
volume_construct_kernel(const float* __restrict__ fL,
                        const float* __restrict__ fR,
                        const float* __restrict__ gamma,
                        const float* __restrict__ beta,
                        const float* __restrict__ mean,
                        const float* __restrict__ var,
                        const float* __restrict__ convw,
                        const float* __restrict__ convb,
                        float* __restrict__ out)
{
    __shared__ __attribute__((aligned(16))) unsigned char smem[SMEM_TOT];
    float*          sScale = (float*)(smem + OFF_SCALE);
    float*          sShift = (float*)(smem + OFF_SHIFT);
    float*          sBias  = (float*)(smem + OFF_BIAS);
    unsigned short* sWb    = (unsigned short*)(smem + OFF_WB);
    unsigned short* sFL    = (unsigned short*)(smem + OFF_FL);
    unsigned short* sFR    = (unsigned short*)(smem + OFF_FR);
    unsigned short* sX     = (unsigned short*)(smem + OFF_X);
    float*          sVol   = (float*)(smem + OFF_X);   // overlays sX

    const int tid  = threadIdx.x;
    // Wave id as an SGPR: loops below become scalar-branch loops, so EXEC is
    // provably all-ones at every WMMA (ISA requirement) and no per-iteration
    // exec-mask bookkeeping is generated.
    const int wv   = __builtin_amdgcn_readfirstlane(tid >> 5);
    const int lane = tid & 31;
    const int n16  = lane & 15;    // N index (and A-matrix M index)
    const int hi   = lane >> 4;    // lane-half selector for K split

    const int b = blockIdx.x / HH;
    const int h = blockIdx.x % HH;

    // ---- Phase 0: BN affine params, bias, bf16 conv weights (K-padded) ----
    if (tid < CIN) {
        float s = gamma[tid] * rsqrtf(var[tid] + 1e-5f);
        sScale[tid] = s;
        sShift[tid] = beta[tid] - mean[tid] * s;
    }
    if (tid < COUT) sBias[tid] = convb[tid];
    for (int idx = tid; idx < COUT * 96; idx += 256) {
        int o = idx / 96, c = idx % 96;
        float v = (c < CIN) ? convw[o * CIN + c] : 0.0f;
        sWb[o * XS + c] = f32_to_bf16(v);
    }
    __syncthreads();

    const size_t ibase = ((size_t)b * CIN * HH + h) * WW;   // + c*HH*WW + w

    // ---- Per image: stage (ReLU+BN -> bf16, transposed) then WMMA conv ----
    for (int img = 0; img < 2; ++img) {
        const float* src = img ? fR : fL;
        unsigned short* feat = img ? sFR : sFL;

        for (int idx = tid; idx < CIN * WW; idx += 256) {
            int c = idx / WW, w = idx % WW;           // coalesced along w
            float x = __builtin_nontemporal_load(src + ibase + (size_t)c * (HH * WW) + w);
            x = fmaxf(x, 0.0f);
            x = x * sScale[c] + sShift[c];
            sX[w * XS + c] = f32_to_bf16(x);
        }
        for (int idx = tid; idx < 16 * WW; idx += 256) {   // zero K-pad 80..95
            int w = idx >> 4, c = CIN + (idx & 15);
            sX[w * XS + c] = 0;
        }
        __syncthreads();

        // conv: out[32 x 320] = Wb[32 x 96] * X[96 x 320], K in 3 steps of 32
        {
            const int mt = wv >> 2;          // waves 0-3 -> out ch 0-15; 4-7 -> 16-31
            const int o0 = mt * 16;

            v16bf afr[3];
#pragma unroll
            for (int ks = 0; ks < 3; ++ks) {
                int kk = ks * 32;
                const unsigned char* rowp = (const unsigned char*)(sWb + (o0 + n16) * XS);
                v4u c0 = *(const v4u*)(rowp + (kk + 16 * hi) * 2);        // K = kk+16hi..+7
                v4u c1 = *(const v4u*)(rowp + (kk + 16 * hi + 8) * 2);    // K = kk+16hi+8..+15
                afr[ks] = frag_cast(c0, c1);
            }

            // Bias for this lane's 8 D elements (o = o0 + 8hi + r), hoisted.
            float biasr[8];
#pragma unroll
            for (int r = 0; r < 8; ++r) biasr[r] = sBias[o0 + 8 * hi + r];

            for (int nt = (wv & 3); nt < 20; nt += 4) {   // scalar loop
                int w0 = nt * 16;
                v8f acc = {};
#pragma unroll
                for (int ks = 0; ks < 3; ++ks) {
                    int kk = ks * 32;
                    const unsigned char* rowp = (const unsigned char*)(sX + (w0 + n16) * XS);
                    v4u c0 = *(const v4u*)(rowp + (kk + 16 * hi) * 2);       // K = kk+16hi..+15
                    v4u c1 = *(const v4u*)(rowp + (kk + 16 * hi) * 2 + 16);
                    v16bf bfr = frag_cast(c0, c1);
                    acc = __builtin_amdgcn_wmma_f32_16x16x32_bf16(
                        false, afr[ks], false, bfr, (short)0, acc, false, false);
                }
                // D[m,n]: m = out-ch (r + 8hi), n = w. Add bias, pack 8 bf16,
                // store transposed [w][c] as one 16B chunk.
                unsigned int up[4];
#pragma unroll
                for (int r = 0; r < 4; ++r) {
                    float vlo = acc[2 * r]     + biasr[2 * r];
                    float vhi = acc[2 * r + 1] + biasr[2 * r + 1];
                    up[r] = (unsigned int)f32_to_bf16(vlo) |
                            ((unsigned int)f32_to_bf16(vhi) << 16);
                }
                v4u u4 = {up[0], up[1], up[2], up[3]};
                *(v4u*)((unsigned char*)(feat + (w0 + n16) * CS) + (o0 + 8 * hi) * 2) = u4;
            }
        }
        __syncthreads();   // feat done; sX may be reused (next img / volume)
    }

    // ---- Phase 3: banded Gram. G[w,w'] = sum_c L[w,c] R[w',c]; i = w - w'.
    // 20 w-tiles x 4 shift-tiles = 80 tiles, one K=32 WMMA each. Scalar loop.
    for (int id = wv; id < 80; id += 8) {
        int wt = id >> 2, t = id & 3;
        int w0 = wt * 16;

        const unsigned char* arow = (const unsigned char*)(sFL + (w0 + n16) * CS);
        v4u a0 = *(const v4u*)(arow + 32 * hi);
        v4u a1 = *(const v4u*)(arow + 32 * hi + 16);
        v16bf afr = frag_cast(a0, a1);

        int wp = w0 - 16 * t + n16;           // w' column for this lane
        v4u b0 = {0, 0, 0, 0}, b1 = {0, 0, 0, 0};
        if (wp >= 0) {                        // zero-pad: shift beyond left edge
            const unsigned char* brow = (const unsigned char*)(sFR + wp * CS);
            b0 = *(const v4u*)(brow + 32 * hi);
            b1 = *(const v4u*)(brow + 32 * hi + 16);
        }
        v16bf bfr = frag_cast(b0, b1);

        v8f acc = {};
        acc = __builtin_amdgcn_wmma_f32_16x16x32_bf16(
            false, afr, false, bfr, (short)0, acc, false, false);

        // D[m,n] = G[w0+m, w0-16t+n]; disparity i = m - n + 16t
#pragma unroll
        for (int r = 0; r < 8; ++r) {
            int m = r + 8 * hi;
            int i = m - n16 + 16 * t;
            if (i >= 0 && i < DD)
                sVol[i * VSTR + w0 + m] = acc[r] * (1.0f / 32.0f);
        }
    }
    __syncthreads();

    // ---- Phase 4: b128 coalesced nontemporal writeback of the (b,h) row ----
    const size_t obase = ((size_t)b * DD * HH + h) * WW;
    for (int idx = tid; idx < DD * (WW / 4); idx += 256) {
        int i = idx / (WW / 4);
        int w = (idx % (WW / 4)) * 4;
        v4f v = *(const v4f*)(sVol + i * VSTR + w);       // 16B aligned (VSTR%4==0)
        __builtin_nontemporal_store(v, (v4f*)(out + obase + (size_t)i * (HH * WW) + w));
    }
}

extern "C" void kernel_launch(void* const* d_in, const int* in_sizes, int n_in,
                              void* d_out, int out_size, void* d_ws, size_t ws_size,
                              hipStream_t stream) {
    (void)in_sizes; (void)n_in; (void)out_size; (void)d_ws; (void)ws_size;
    const float* fL    = (const float*)d_in[0];
    const float* fR    = (const float*)d_in[1];
    const float* gamma = (const float*)d_in[2];
    const float* beta  = (const float*)d_in[3];
    const float* mean  = (const float*)d_in[4];
    const float* var   = (const float*)d_in[5];
    const float* convw = (const float*)d_in[6];
    const float* convb = (const float*)d_in[7];
    float* out = (float*)d_out;

    dim3 grid(BB * HH);   // 1280 row-blocks
    dim3 block(256);      // 8 wave32
    volume_construct_kernel<<<grid, block, 0, stream>>>(
        fL, fR, gamma, beta, mean, var, convw, convb, out);
}
